// MultiheadAttentionPooling_12481174962242
// MI455X (gfx1250) — compile-verified, matrix operand order checked by reading, not run on hardware
//
#include <hip/hip_runtime.h>
#include <hip/hip_bf16.h>

// Sizes from the reference
#define BB 32
#define NN 2048
#define DD 1024
#define HH 16
#define HD 64

typedef __attribute__((ext_vector_type(16))) _Float16 v16h;
typedef __attribute__((ext_vector_type(8)))  float    v8f;
typedef __attribute__((ext_vector_type(4)))  float    v4f;

// ---------------------------------------------------------------------------
// K0: q[j] = query . w_q[j,:] + b_q[j]     (1x1024 @ 1024x1024^T, tiny)
// ---------------------------------------------------------------------------
__global__ void k_qproj(const float* __restrict__ query, const float* __restrict__ w_q,
                        const float* __restrict__ b_q, float* __restrict__ q) {
  __shared__ float sq[DD];
  for (int i = threadIdx.x; i < DD; i += blockDim.x) sq[i] = query[i];
  __syncthreads();
  int j = blockIdx.x * blockDim.x + threadIdx.x;
  if (j < DD) {
    float acc = 0.f;
    const float* wr = w_q + (size_t)j * DD;
    for (int d = 0; d < DD; ++d) acc = fmaf(sq[d], wr[d], acc);
    q[j] = acc + b_q[j];
  }
}

// ---------------------------------------------------------------------------
// K_mask: has_valid[b] = OR over mask row (for the "force column 0 valid" rule)
// ---------------------------------------------------------------------------
__global__ void k_hasvalid(const unsigned char* __restrict__ mask, int* __restrict__ hasvalid) {
  int b = blockIdx.x;
  __shared__ int s;
  if (threadIdx.x == 0) s = 0;
  __syncthreads();
  int any = 0;
  for (int n = threadIdx.x; n < NN; n += blockDim.x) any |= (int)mask[b * NN + n];
  if (any) atomicOr(&s, 1);
  __syncthreads();
  if (threadIdx.x == 0) hasvalid[b] = s;
}

// ---------------------------------------------------------------------------
// K1: qk[h,d] = sum_{j in head h} q[j] * w_k[j,d], stored as f16 directly in
// the per-lane WMMA B-operand layout:
//   qkw[kb*512 + lane*16 + i]  for k-block kb (K0 = 32*kb), 32 lanes, 16 halfs.
//   lane = col h (lane&15) ; lane<16 -> K in {0..7,16..23}, else {8..15,24..31}
// (mirrors the documented 16-bit A-matrix 16x32 K ordering per lane group)
// ---------------------------------------------------------------------------
__global__ void k_qk(const float* __restrict__ q, const float* __restrict__ w_k,
                     _Float16* __restrict__ qkw) {
  int t = blockIdx.x * blockDim.x + threadIdx.x;  // 0..16383
  int kb   = t >> 9;
  int lane = (t >> 4) & 31;
  int i    = t & 15;
  int h    = lane & 15;
  int base = (lane < 16) ? 0 : 8;
  int klocal = (i < 8) ? (base + i) : (16 + base + (i - 8));
  int d = kb * 32 + klocal;
  float acc = 0.f;
  const float* qh = q + h * HD;
  const float* wk = w_k + (size_t)(h * HD) * DD + d;
#pragma unroll 8
  for (int jj = 0; jj < HD; ++jj) acc = fmaf(qh[jj], wk[(size_t)jj * DD], acc);
  qkw[t] = (_Float16)acc;
}

// ---------------------------------------------------------------------------
// K2: scores[b,h,n] = scale * (z[b,n,:] . qk[h,:]), masked to -inf.
// One wave per 16-row tile of z; N-dim = 16 heads. K-loop over D with
// v_wmma_f32_16x16x32_f16. 8 waves (8 tiles) per 256-thread block.
// ---------------------------------------------------------------------------
__global__ void k_scores(const float* __restrict__ z, const _Float16* __restrict__ qkw,
                         const unsigned char* __restrict__ mask,
                         const int* __restrict__ hasvalid,
                         float* __restrict__ scores) {
  const int lane = threadIdx.x & 31;
  const int tile = blockIdx.x * 8 + (threadIdx.x >> 5);   // 16 rows each
  const long m0 = (long)tile * 16;
  const int b  = (int)(m0 / NN);
  const int n0 = (int)(m0 % NN);
  const int arow = (int)m0 + (lane & 15);
  const int koff = (lane < 16) ? 0 : 8;
  const float* zr = z + (size_t)arow * DD;

  v8f c = {};
#pragma unroll 4
  for (int kb = 0; kb < DD / 32; ++kb) {
    const int k0 = kb * 32 + koff;
    v4f f0 = *(const v4f*)(zr + k0);
    v4f f1 = *(const v4f*)(zr + k0 + 4);
    v4f f2 = *(const v4f*)(zr + k0 + 16);
    v4f f3 = *(const v4f*)(zr + k0 + 20);
    v16h a;
#pragma unroll
    for (int e = 0; e < 4; ++e) {
      a[e]      = (_Float16)f0[e];
      a[4 + e]  = (_Float16)f1[e];
      a[8 + e]  = (_Float16)f2[e];
      a[12 + e] = (_Float16)f3[e];
    }
    v16h bm = *(const v16h*)(qkw + kb * 512 + lane * 16);
    c = __builtin_amdgcn_wmma_f32_16x16x32_f16(false, a, false, bm, (short)0, c,
                                               false, false);
  }

  // D layout: lanes 0-15 -> col=lane, rows r=0..7 ; lanes 16-31 -> rows 8..15
  const float scale = 0.125f;  // 1/sqrt(64)
  const int h = lane & 15;
  const int mbase = (lane < 16) ? 0 : 8;
  const int hv = hasvalid[b];
#pragma unroll
  for (int r = 0; r < 8; ++r) {
    int n = n0 + mbase + r;
    bool valid = (mask[(size_t)b * NN + n] != 0) || ((n == 0) && (hv == 0));
    float val = valid ? c[r] * scale : -__builtin_inff();
    scores[((size_t)(b * HH + h)) * NN + n] = val;
  }
}

// ---------------------------------------------------------------------------
// K3: softmax over N per (b,h) row, in place.
// ---------------------------------------------------------------------------
__global__ void k_softmax(float* __restrict__ scores) {
  float* s = scores + (size_t)blockIdx.x * NN;
  __shared__ float red[256];
  float m = -__builtin_inff();
  for (int n = threadIdx.x; n < NN; n += 256) m = fmaxf(m, s[n]);
  red[threadIdx.x] = m;
  __syncthreads();
  for (int o = 128; o > 0; o >>= 1) {
    if (threadIdx.x < o) red[threadIdx.x] = fmaxf(red[threadIdx.x], red[threadIdx.x + o]);
    __syncthreads();
  }
  m = red[0];
  __syncthreads();
  float sum = 0.f;
  for (int n = threadIdx.x; n < NN; n += 256) {
    float e = __expf(s[n] - m);   // exp(-inf - m) == 0 for masked entries
    s[n] = e;
    sum += e;
  }
  red[threadIdx.x] = sum;
  __syncthreads();
  for (int o = 128; o > 0; o >>= 1) {
    if (threadIdx.x < o) red[threadIdx.x] += red[threadIdx.x + o];
    __syncthreads();
  }
  float inv = 1.f / red[0];
  for (int n = threadIdx.x; n < NN; n += 256) s[n] *= inv;
}

// ---------------------------------------------------------------------------
// K4: zbar[b,h,d] = sum_n attn[b,h,n] * z[b,n,d]   (streaming pass over z)
// block = (b, 256-wide d chunk); attn tile staged in LDS.
// ---------------------------------------------------------------------------
__global__ void k_zbar(const float* __restrict__ z, const float* __restrict__ attn,
                       float* __restrict__ zbar) {
  const int b = blockIdx.x >> 2;
  const int d = ((blockIdx.x & 3) << 8) + threadIdx.x;
  __shared__ float sa[HH][128];
  float acc[HH];
#pragma unroll
  for (int h = 0; h < HH; ++h) acc[h] = 0.f;
  for (int nc = 0; nc < NN; nc += 128) {
    __syncthreads();
    for (int idx = threadIdx.x; idx < HH * 128; idx += 256) {
      int h = idx >> 7, nn = idx & 127;
      sa[h][nn] = attn[((size_t)(b * HH + h)) * NN + nc + nn];
    }
    __syncthreads();
    const float* zp = z + ((size_t)b * NN + nc) * DD + d;
    for (int nn = 0; nn < 128; ++nn) {
      float zv = zp[(size_t)nn * DD];
#pragma unroll
      for (int h = 0; h < HH; ++h) acc[h] = fmaf(sa[h][nn], zv, acc[h]);
    }
  }
#pragma unroll
  for (int h = 0; h < HH; ++h) zbar[((size_t)(b * HH + h)) * DD + d] = acc[h];
}

// ---------------------------------------------------------------------------
// K5: pooled[b,j] = zbar[b, j/64, :] . w_v[j,:] + b_v[j]
// One thread per output column j, all 32 batches in registers.
// ---------------------------------------------------------------------------
__global__ void k_pooled(const float* __restrict__ zbar, const float* __restrict__ w_v,
                         const float* __restrict__ b_v, float* __restrict__ pooled) {
  int j = blockIdx.x * 256 + threadIdx.x;
  int h = j / HD;
  float acc[BB];
#pragma unroll
  for (int bb = 0; bb < BB; ++bb) acc[bb] = 0.f;
  const float* wr = w_v + (size_t)j * DD;
  for (int d = 0; d < DD; ++d) {
    float wv = wr[d];
#pragma unroll
    for (int bb = 0; bb < BB; ++bb)
      acc[bb] = fmaf(zbar[((size_t)(bb * HH + h)) * DD + d], wv, acc[bb]);
  }
#pragma unroll
  for (int bb = 0; bb < BB; ++bb) pooled[(size_t)bb * DD + j] = acc[bb] + b_v[j];
}

// ---------------------------------------------------------------------------
// K6: out[b,i] = pooled[b,:] . w_o[i,:] + b_o[i]
// ---------------------------------------------------------------------------
__global__ void k_out(const float* __restrict__ pooled, const float* __restrict__ w_o,
                      const float* __restrict__ b_o, float* __restrict__ out) {
  int i = blockIdx.x * 256 + threadIdx.x;
  float acc[BB];
#pragma unroll
  for (int bb = 0; bb < BB; ++bb) acc[bb] = 0.f;
  const float* wr = w_o + (size_t)i * DD;
  for (int j = 0; j < DD; ++j) {
    float wv = wr[j];
#pragma unroll
    for (int bb = 0; bb < BB; ++bb)
      acc[bb] = fmaf(pooled[(size_t)bb * DD + j], wv, acc[bb]);
  }
#pragma unroll
  for (int bb = 0; bb < BB; ++bb) out[(size_t)bb * DD + i] = acc[bb] + b_o[i];
}

// ---------------------------------------------------------------------------
extern "C" void kernel_launch(void* const* d_in, const int* in_sizes, int n_in,
                              void* d_out, int out_size, void* d_ws, size_t ws_size,
                              hipStream_t stream) {
  const float*         z     = (const float*)d_in[0];
  const unsigned char* mask  = (const unsigned char*)d_in[1];  // jnp bool -> 1 byte
  const float*         query = (const float*)d_in[2];
  const float*         w_q   = (const float*)d_in[3];
  const float*         w_k   = (const float*)d_in[4];
  const float*         w_v   = (const float*)d_in[5];
  const float*         b_q   = (const float*)d_in[6];
  // d_in[7] = b_k: a per-head constant shift of scores -> softmax invariant, dropped.
  const float*         b_v   = (const float*)d_in[8];
  const float*         w_o   = (const float*)d_in[9];
  const float*         b_o   = (const float*)d_in[10];
  float*               out   = (float*)d_out;

  char* ws = (char*)d_ws;
  float*    q        = (float*)(ws + 0);                         // 4 KB
  int*      hasvalid = (int*)(ws + 4096);                        // 128 B
  _Float16* qkw      = (_Float16*)(ws + 8192);                   // 32 KB
  float*    scores   = (float*)(ws + 65536);                     // 4 MB (-> attn)
  float*    zbar     = (float*)(ws + 65536 + 4 * 1024 * 1024);   // 2 MB
  float*    pooled   = (float*)(ws + 65536 + 6 * 1024 * 1024);   // 128 KB

  k_qproj   <<<4, 256, 0, stream>>>(query, w_q, b_q, q);
  k_hasvalid<<<BB, 256, 0, stream>>>(mask, hasvalid);
  k_qk      <<<64, 256, 0, stream>>>(q, w_k, qkw);
  k_scores  <<<(BB * NN) / 16 / 8, 256, 0, stream>>>(z, qkw, mask, hasvalid, scores);
  k_softmax <<<BB * HH, 256, 0, stream>>>(scores);
  k_zbar    <<<BB * 4, 256, 0, stream>>>(z, scores, zbar);
  k_pooled  <<<4, 256, 0, stream>>>(zbar, w_v, b_v, pooled);
  k_out     <<<4, 256, 0, stream>>>(pooled, w_o, b_o, out);
}
